// SwinFuncSelfAttention_15436112462237
// MI455X (gfx1250) — matrix operand order repaired and checked by reading, not compiled, not run on hardware
//
#include <hip/hip_runtime.h>

typedef __attribute__((ext_vector_type(16))) _Float16 v16h;
typedef __attribute__((ext_vector_type(8)))  _Float16 v8h;
typedef __attribute__((ext_vector_type(8)))  float    v8f;
typedef __attribute__((ext_vector_type(4)))  float    v4f;

#define BN   64
#define S_   64
#define C_   128
#define H_   8
#define PP_  64      // p*p
#define KD_  1024    // hd * p * p   (labeled kdim = yx*16 + d)

// ---- workspace layout (bytes) ----
#define OFF_BIAS 0UL
#define SZ_BIAS  (4UL*8*64*64*4)                 // 512 KB  f32 [w][h][q][k]
#define OFF_Q    (OFF_BIAS + SZ_BIAS)
#define SZ_QKV   (64UL*8*64*1024*2)              // 64 MB each
#define OFF_K    (OFF_Q  + SZ_QKV)
#define OFF_VROW (OFF_K  + SZ_QKV)               // V row-major [bh][s][n]
#define OFF_VT   (OFF_VROW + SZ_QKV)             // V transposed [bh][n][s]
#define OFF_OT   OFF_VROW                        // Ot aliases dead Vrow

// cyclic shift map: subdomain g in [0,16) -> (g+4) mod 16   (same map for
// the forward -shift gather and the final +shift scatter)
__device__ __forceinline__ void shiftmap(int w, int s8, int& w2, int& s2) {
    int g = ((w << 3) + s8 + 4) & 15;
    w2 = g >> 3; s2 = g & 7;
}

// two contiguous 16B chunks -> one v16h fragment
__device__ __forceinline__ v16h ld_pair(const _Float16* p, int off2) {
    v8h lo = *(const v8h*)p;
    v8h hi = *(const v8h*)(p + off2);
    return __builtin_shufflevector(lo, hi, 0,1,2,3,4,5,6,7,8,9,10,11,12,13,14,15);
}

// fragment built from fp32 source with conversion (weights, loaded once)
__device__ __forceinline__ v16h ld_frag_f32(const float* p, int off2) {
    v16h a;
#pragma unroll
    for (int i = 0; i < 8; ++i) a[i]     = (_Float16)p[i];
#pragma unroll
    for (int i = 0; i < 8; ++i) a[i + 8] = (_Float16)p[off2 + i];
    return a;
}

__device__ __forceinline__ int region16(int g) { return g < 8 ? 0 : (g < 12 ? 1 : 2); }

// ---------------------------------------------------------------------------
// Kernel 1: bias table = shifted-window mask + SwinV2 log-CPB MLP
// ---------------------------------------------------------------------------
__global__ void __launch_bounds__(256) k_bias(const float* __restrict__ w1,
                                              const float* __restrict__ b1,
                                              const float* __restrict__ w2,
                                              const float* __restrict__ b2,
                                              float* __restrict__ biasBuf) {
    int idx = blockIdx.x * blockDim.x + threadIdx.x;
    if (idx >= 64 * 64) return;
    int i = idx >> 6, j = idx & 63;
    int iy = i >> 3, ix = i & 7, jy = j >> 3, jx = j & 7;

    float ry = (float)(iy - jy) * (8.0f / 7.0f);
    float rx = (float)(ix - jx) * (8.0f / 7.0f);
    float ly = log2f(fabsf(ry) + 1.0f) * (1.0f / 3.0f);
    float lx = log2f(fabsf(rx) + 1.0f) * (1.0f / 3.0f);
    ry = ry < 0.f ? -ly : ly;
    rx = rx < 0.f ? -lx : lx;

    float acc[8] = {0.f,0.f,0.f,0.f,0.f,0.f,0.f,0.f};
    for (int u = 0; u < 512; ++u) {
        float hv = fmaxf(ry * w1[u] + rx * w1[512 + u] + b1[u], 0.0f);
#pragma unroll
        for (int h = 0; h < 8; ++h) acc[h] += hv * w2[u * 8 + h];
    }
#pragma unroll
    for (int h = 0; h < 8; ++h) {
        float cpb = 16.0f / (1.0f + __expf(-(acc[h] + b2[h])));
#pragma unroll
        for (int w = 0; w < 4; ++w) {
            int wy = w >> 1, wx = w & 1;
            int vi = 3 * region16(wy * 8 + iy) + region16(wx * 8 + ix);
            int vj = 3 * region16(wy * 8 + jy) + region16(wx * 8 + jx);
            float m = (vi != vj) ? -1e9f : 0.0f;
            biasBuf[(((size_t)(w * 8 + h) * 64 + i) * 64) + j] = m + cpb;
        }
    }
}

// ---------------------------------------------------------------------------
// Kernel 2: shift-gather + QKV projection.  block = bn (64 blocks, 8 waves).
// Wave w owns head w; g=0/1/2 selects the Q/K/V slice of qkv_w.
// All outputs stored row-major [bh][s][n], n = yx*16+d  ->  b128 stores.
// ---------------------------------------------------------------------------
__global__ void __launch_bounds__(256) k_qkv(const float* __restrict__ seq,
                                             const float* __restrict__ qkv_w,
                                             _Float16* __restrict__ Q,
                                             _Float16* __restrict__ K,
                                             _Float16* __restrict__ Vrow) {
    __shared__ __align__(16) _Float16 Xt[64 * 128];   // [yx][c], 16 KB

    int bn = blockIdx.x, t = threadIdx.x;
    int lane = t & 31, wv = t >> 5, l15 = lane & 15, half = lane >> 4;

    // hoist W fragments: mt = wv + 8g  ->  (g: q/k/v slice) x (head wv) rows
    v16h A[3][4];
#pragma unroll
    for (int g = 0; g < 3; ++g) {
        int mt = wv + 8 * g;
#pragma unroll
        for (int kc = 0; kc < 4; ++kc)
            A[g][kc] = ld_frag_f32(qkv_w + (mt * 16 + l15) * 128 + kc * 32 + half * 8, 16);
    }

    int b = bn >> 2, wy = (bn >> 1) & 1, wx = bn & 1;
    size_t hb = (size_t)(bn * 8 + wv);

    for (int s = 0; s < 64; ++s) {
        int wy2, sy2, wx2, sx2;
        shiftmap(wy, s >> 3, wy2, sy2);
        shiftmap(wx, s & 7, wx2, sx2);
        const float* xsrc = seq + ((size_t)((b * 4 + wy2 * 2 + wx2) * 64 + sy2 * 8 + sx2)) * C_ * PP_;

        __syncthreads();                                // prev iteration done reading Xt
        for (int i = t; i < 128 * 64; i += 256) {
            int c = i >> 6, yx = i & 63;
            Xt[yx * 128 + c] = (_Float16)xsrc[i];
        }
        __syncthreads();

#pragma unroll
        for (int nt = 0; nt < 4; ++nt) {
            v16h B[4];
#pragma unroll
            for (int kc = 0; kc < 4; ++kc)
                B[kc] = ld_pair(&Xt[(nt * 16 + l15) * 128 + kc * 32 + half * 8], 16);
            // destination: n = yx*16 + d, lane's 8 outputs d = 8*half + r contiguous
            size_t dofs = (hb * 64 + s) * 1024 + (size_t)(nt * 16 + l15) * 16 + 8 * half;
#pragma unroll
            for (int g = 0; g < 3; ++g) {
                v8f acc = {};
#pragma unroll
                for (int kc = 0; kc < 4; ++kc)
                    acc = __builtin_amdgcn_wmma_f32_16x16x32_f16(false, A[g][kc], false, B[kc],
                                                                 (short)0, acc, false, false);
                v8h o8;
#pragma unroll
                for (int r = 0; r < 8; ++r) o8[r] = (_Float16)acc[r];
                _Float16* dst = (g == 0) ? Q : (g == 1) ? K : Vrow;
                *(v8h*)(dst + dofs) = o8;
            }
        }
    }
}

// ---------------------------------------------------------------------------
// Kernel 2.5: V transpose  [bh][s=64][n=1024] -> [bh][n][s]  via LDS tiles.
// block = (bn,h), 256 threads, 16 chunks of 64 n.
// ---------------------------------------------------------------------------
__global__ void __launch_bounds__(256) k_vt(const _Float16* __restrict__ Vrow,
                                            _Float16* __restrict__ Vt) {
    __shared__ __align__(16) _Float16 T[64 * 64];     // 8 KB tile [s][n]
    int bh = blockIdx.x, t = threadIdx.x;
    const uint4* src = (const uint4*)(Vrow + (size_t)bh * 64 * 1024);
    v8h* dst = (v8h*)(Vt + (size_t)bh * 1024 * 64);
    uint4* Tu = (uint4*)T;

    for (int c = 0; c < 16; ++c) {
        __syncthreads();
#pragma unroll
        for (int u = t; u < 512; u += 256) {
            int s = u >> 3, n8 = u & 7;
            Tu[u] = src[s * 128 + c * 8 + n8];        // coalesced b128 loads
        }
        __syncthreads();
#pragma unroll
        for (int u = t; u < 512; u += 256) {
            int n = u >> 3, s8 = u & 7;
            v8h o;
#pragma unroll
            for (int i = 0; i < 8; ++i) o[i] = T[(s8 * 8 + i) * 64 + n];
            dst[(c * 64 + n) * 8 + s8] = o;           // coalesced b128 stores
        }
    }
}

// ---------------------------------------------------------------------------
// Kernel 3: attention.  block = (bn, h), one wave.
// scores = QK^T (512 wmma) -> LDS softmax (+bias) -> oT = Vt * attnT (512 wmma)
// ---------------------------------------------------------------------------
__global__ void __launch_bounds__(32) k_attn(const _Float16* __restrict__ Q,
                                             const _Float16* __restrict__ K,
                                             const _Float16* __restrict__ Vt,
                                             const float* __restrict__ biasBuf,
                                             _Float16* __restrict__ Ot) {
    __shared__ __align__(16) float    sc[64 * 64];   // 16 KB
    __shared__ __align__(16) _Float16 at[64 * 64];   //  8 KB  attn[q][kv]

    int bn = blockIdx.x >> 3, h = blockIdx.x & 7;
    int lane = threadIdx.x, l15 = lane & 15, half = lane >> 4;

    const _Float16* Qb = Q  + (size_t)(bn * 8 + h) * 64 * 1024;
    const _Float16* Kb = K  + (size_t)(bn * 8 + h) * 64 * 1024;
    const _Float16* Vb = Vt + (size_t)(bn * 8 + h) * 1024 * 64;
    const float*    bb = biasBuf + (size_t)((bn & 3) * 8 + h) * 64 * 64;
    const float scale = 1.0f / 256.0f;               // 1/(p*p*sqrt(hd))

    // ---- scores = Q * K^T ----
    for (int mt = 0; mt < 4; ++mt) {
        v8f acc[4] = {{}, {}, {}, {}};
        for (int kc = 0; kc < 32; ++kc) {
            v16h Af = ld_pair(Qb + (mt * 16 + l15) * 1024 + kc * 32 + half * 8, 16);
#pragma unroll
            for (int nt = 0; nt < 4; ++nt) {
                v16h Bf = ld_pair(Kb + (nt * 16 + l15) * 1024 + kc * 32 + half * 8, 16);
                acc[nt] = __builtin_amdgcn_wmma_f32_16x16x32_f16(false, Af, false, Bf,
                                                                 (short)0, acc[nt], false, false);
            }
        }
#pragma unroll
        for (int nt = 0; nt < 4; ++nt)
#pragma unroll
            for (int r = 0; r < 8; ++r)
                sc[(mt * 16 + r + 8 * half) * 64 + nt * 16 + l15] = acc[nt][r] * scale;
    }
    __syncthreads();

    // ---- softmax(scores + bias), 2 rows per lane ----
#pragma unroll
    for (int rr = 0; rr < 2; ++rr) {
        int row = lane + rr * 32;
        const float* brow = bb + row * 64;
        float m = -3.4e38f;
        for (int j = 0; j < 64; ++j) {
            float v = sc[row * 64 + j] + brow[j];
            sc[row * 64 + j] = v;
            m = fmaxf(m, v);
        }
        float sum = 0.f;
        for (int j = 0; j < 64; ++j) {
            float e = __expf(sc[row * 64 + j] - m);
            sc[row * 64 + j] = e;
            sum += e;
        }
        float inv = 1.0f / sum;
        for (int j = 0; j < 64; ++j)
            at[row * 64 + j] = (_Float16)(sc[row * 64 + j] * inv);
    }
    __syncthreads();

    // ---- oT[n,q] = sum_s Vt[n,s] * attn[q,s] ; store Ot[bn][s=q][yx][cin] ----
    // B fragments (attn^T) hoisted: lane = col q, k = s contiguous in at[q][s]
    v16h Bf[4][2];
#pragma unroll
    for (int qt = 0; qt < 4; ++qt)
#pragma unroll
        for (int kc = 0; kc < 2; ++kc)
            Bf[qt][kc] = ld_pair(&at[(qt * 16 + l15) * 64 + kc * 32 + half * 8], 16);

    for (int nt = 0; nt < 64; ++nt) {                 // n-tile: yx = nt, d = 0..15
        v16h A0 = ld_pair(Vb + (nt * 16 + l15) * 64 +  0 + half * 8, 16);
        v16h A1 = ld_pair(Vb + (nt * 16 + l15) * 64 + 32 + half * 8, 16);
#pragma unroll
        for (int qt = 0; qt < 4; ++qt) {
            v8f acc = {};
            acc = __builtin_amdgcn_wmma_f32_16x16x32_f16(false, A0, false, Bf[qt][0], (short)0, acc, false, false);
            acc = __builtin_amdgcn_wmma_f32_16x16x32_f16(false, A1, false, Bf[qt][1], (short)0, acc, false, false);
            int q = qt * 16 + l15;                    // lane column = query row
            v8h o8;                                   // rows n = nt*16 + 8*half + r
#pragma unroll
            for (int r = 0; r < 8; ++r) o8[r] = (_Float16)acc[r];
            *(v8h*)(Ot + ((size_t)(bn * 64 + q) * 64 + nt) * 128 + h * 16 + 8 * half) = o8;
        }
    }
}

// ---------------------------------------------------------------------------
// Kernel 4: output projection + bias + inverse-shift scatter to fp32 out.
// Flipped GEMM: rows = yx (A = Ot), cols = c_out (B = out_w^T, hoisted).
// block = bn (8 waves); wave w owns c_out columns [16w, 16w+16).
// ---------------------------------------------------------------------------
__global__ void __launch_bounds__(256) k_proj(const _Float16* __restrict__ Ot,
                                              const float* __restrict__ out_w,
                                              const float* __restrict__ out_b,
                                              float* __restrict__ out) {
    int bn = blockIdx.x, t = threadIdx.x;
    int lane = t & 31, wv = t >> 5, l15 = lane & 15, half = lane >> 4;

    // B[k=cin, n=c_out] = out_w[c_out][cin]; lane col c_out = wv*16+l15
    v16h Bf[4];
#pragma unroll
    for (int kc = 0; kc < 4; ++kc)
        Bf[kc] = ld_frag_f32(out_w + (wv * 16 + l15) * 128 + kc * 32 + half * 8, 16);
    int c = wv * 16 + l15;                            // this lane's output channel
    float bsc = out_b[c];

    int b = bn >> 2, wy = (bn >> 1) & 1, wx = bn & 1;

    for (int s = 0; s < 64; ++s) {
        int wy2, sy2, wx2, sx2;
        shiftmap(wy, s >> 3, wy2, sy2);
        shiftmap(wx, s & 7, wx2, sx2);
        float* dst = out + (size_t)((b * 4 + wy2 * 2 + wx2) * 64 + sy2 * 8 + sx2) * C_ * PP_;
        const _Float16* Ob = Ot + (size_t)(bn * 64 + s) * 64 * 128;

#pragma unroll
        for (int mt = 0; mt < 4; ++mt) {
            v8f acc = {};
#pragma unroll
            for (int kc = 0; kc < 4; ++kc) {
                v16h A = ld_pair(Ob + (mt * 16 + l15) * 128 + kc * 32 + half * 8, 16);
                acc = __builtin_amdgcn_wmma_f32_16x16x32_f16(false, A, false, Bf[kc],
                                                             (short)0, acc, false, false);
            }
            // rows yx = mt*16 + 8*half + r contiguous -> two float4 stores
            float* dp = dst + c * 64 + mt * 16 + 8 * half;
            v4f lo = {acc[0] + bsc, acc[1] + bsc, acc[2] + bsc, acc[3] + bsc};
            v4f hi = {acc[4] + bsc, acc[5] + bsc, acc[6] + bsc, acc[7] + bsc};
            *(v4f*)dp = lo;
            *(v4f*)(dp + 4) = hi;
        }
    }
}

// ---------------------------------------------------------------------------
extern "C" void kernel_launch(void* const* d_in, const int* in_sizes, int n_in,
                              void* d_out, int out_size, void* d_ws, size_t ws_size,
                              hipStream_t stream) {
    const float* seq    = (const float*)d_in[0];
    const float* qkv_w  = (const float*)d_in[1];
    const float* out_w  = (const float*)d_in[2];
    const float* out_b  = (const float*)d_in[3];
    const float* cpb_w1 = (const float*)d_in[4];
    const float* cpb_b1 = (const float*)d_in[5];
    const float* cpb_w2 = (const float*)d_in[6];
    const float* cpb_b2 = (const float*)d_in[7];

    char* ws = (char*)d_ws;
    float*    bias = (float*)   (ws + OFF_BIAS);
    _Float16* Q    = (_Float16*)(ws + OFF_Q);
    _Float16* K    = (_Float16*)(ws + OFF_K);
    _Float16* Vrow = (_Float16*)(ws + OFF_VROW);
    _Float16* Vt   = (_Float16*)(ws + OFF_VT);
    _Float16* Ot   = (_Float16*)(ws + OFF_OT);       // aliases Vrow (dead by then)

    k_bias<<<16, 256, 0, stream>>>(cpb_w1, cpb_b1, cpb_w2, cpb_b2, bias);
    k_qkv <<<64, 256, 0, stream>>>(seq, qkv_w, Q, K, Vrow);
    k_vt  <<<512, 256, 0, stream>>>(Vrow, Vt);
    k_attn<<<512, 32, 0, stream>>>(Q, K, Vt, bias, Ot);
    k_proj<<<64, 256, 0, stream>>>(Ot, out_w, out_b, (float*)d_out);
}
